// multiHeadAttention_52639119180072
// MI455X (gfx1250) — matrix operand order repaired and checked by reading, not compiled, not run on hardware
//
#include <hip/hip_runtime.h>
#include <stdint.h>
#include <stddef.h>

// ---------------------------------------------------------------------------
// MHA forward for MI455X (gfx1250, wave32, WMMA).
// bf16 operands + f32 accumulation via v_wmma_f32_16x16x32_bf16.
// HBM staging through GLOBAL_LOAD_ASYNC_TO_LDS_B128 (ASYNCcnt) when the
// toolchain exposes the builtin; double-buffered K-loop in the GEMMs.
// ---------------------------------------------------------------------------

#define B_  32
#define N_  577
#define E_  768
#define H_  12
#define D_  64
#define M_  (B_ * N_)          // 18464 rows

typedef __attribute__((ext_vector_type(16))) __bf16 v16bf;
typedef __attribute__((ext_vector_type(8)))  float  v8f;

union Frag { v16bf v; uint4 q[2]; };

__device__ __forceinline__ uint16_t f2bf(float f) {
    uint32_t u = __builtin_bit_cast(uint32_t, f);
    uint32_t r = (u + 0x7FFFu + ((u >> 16) & 1u)) >> 16;   // RNE
    return (uint16_t)r;
}
__device__ __forceinline__ float bf2f(uint16_t b) {
    uint32_t u = ((uint32_t)b) << 16;
    return __builtin_bit_cast(float, u);
}

// ---- 16B global -> LDS copy: async (ASYNCcnt) when available --------------
#if __has_builtin(__builtin_amdgcn_global_load_async_to_lds_b128)
#define ASYNC_LDS 1
typedef int i32x4_t __attribute__((vector_size(16)));
typedef __attribute__((address_space(1))) i32x4_t* gptr_i32x4;
typedef __attribute__((address_space(3))) i32x4_t* lptr_i32x4;
__device__ __forceinline__ void copy16(const uint16_t* g, uint16_t* l) {
    __builtin_amdgcn_global_load_async_to_lds_b128(
        (gptr_i32x4)g, (lptr_i32x4)l, 0, 0);
}
__device__ __forceinline__ void async_fence() {
    asm volatile("s_wait_asynccnt 0" ::: "memory");
}
#else
#define ASYNC_LDS 0
__device__ __forceinline__ void copy16(const uint16_t* g, uint16_t* l) {
    *(uint4*)l = *(const uint4*)g;
}
__device__ __forceinline__ void async_fence() {}
#endif

// ---------------------------------------------------------------------------
// fp32 -> bf16 conversion (one HBM round trip per tensor)
// ---------------------------------------------------------------------------
__global__ void f32_to_bf16_kernel(const float* __restrict__ src,
                                   uint16_t* __restrict__ dst, int count) {
    int i = blockIdx.x * blockDim.x + threadIdx.x;
    if (i < count) dst[i] = f2bf(src[i]);
}

// ---------------------------------------------------------------------------
// Tiled bf16 WMMA GEMM:  C[M,N] = A[M,768] * B[N,768]^T + bias
// Block tile 128x64, K-step 64, double-buffered async LDS staging,
// 8 waves (4 M x 2 N), 32x32 per wave (2x2 WMMA accumulators).
// MODE 0: fused QKV projection, scatter-store bf16 into [b,h,n,d] Q/K/V.
// MODE 1: output projection, fp32 store to d_out.
// ---------------------------------------------------------------------------
template <int MODE>
__global__ __launch_bounds__(256) void gemm_bf16_kernel(
    const uint16_t* __restrict__ A,    // [Mrows, 768] bf16
    const uint16_t* __restrict__ Bw,   // [Ntot , 768] bf16 (row = out col)
    const float* __restrict__ bias0, const float* __restrict__ bias1,
    const float* __restrict__ bias2,
    uint16_t* __restrict__ Qb, uint16_t* __restrict__ Kb,
    uint16_t* __restrict__ Vb,
    float* __restrict__ Out, int Mrows) {
    __shared__ __align__(16) uint16_t As[2][128 * 64];   // 2 x 16 KB
    __shared__ __align__(16) uint16_t Bs[2][64 * 64];    // 2 x  8 KB
    const int tid    = threadIdx.x;
    const int lane   = tid & 31;
    const int w      = tid >> 5;
    const int wm     = w & 3, wn = w >> 2;
    const int laneHi = lane >> 4, l15 = lane & 15;
    const int bm0 = blockIdx.x * 128, bn0 = blockIdx.y * 64;

    v8f acc[2][2] = {};

    auto stage = [&](int buf, int k0) {
        #pragma unroll
        for (int c = tid; c < 1024; c += 256) {          // A: 128x64
            int row = c >> 3, seg = c & 7;
            int gm = bm0 + row; if (gm >= Mrows) gm = Mrows - 1;
            copy16(A + (size_t)gm * 768 + k0 + seg * 8,
                   &As[buf][row * 64 + seg * 8]);
        }
        #pragma unroll
        for (int c = tid; c < 512; c += 256) {           // B: 64x64
            int row = c >> 3, seg = c & 7;
            copy16(Bw + (size_t)(bn0 + row) * 768 + k0 + seg * 8,
                   &Bs[buf][row * 64 + seg * 8]);
        }
    };

    stage(0, 0);
    async_fence();
    __syncthreads();

    for (int k0 = 0; k0 < 768; k0 += 64) {
        const int buf = (k0 >> 6) & 1;
        if (k0 + 64 < 768) stage(buf ^ 1, k0 + 64);   // overlap next stage

        #pragma unroll
        for (int kk = 0; kk < 64; kk += 32) {
            Frag af[2], bfr[2];
            #pragma unroll
            for (int mi = 0; mi < 2; mi++) {
                // A frag: lane<16 row L holds K {0..7,16..23}; lane>=16 +8
                int row = wm * 32 + mi * 16 + l15;
                int e0 = kk + (laneHi ? 8 : 0);
                af[mi].q[0] = *(const uint4*)(&As[buf][row * 64 + e0]);
                af[mi].q[1] = *(const uint4*)(&As[buf][row * 64 + e0 + 16]);
            }
            #pragma unroll
            for (int ni = 0; ni < 2; ni++) {
                // B frag: lane<16 col c holds K {0..7,16..23}; lane>=16 +8
                int col = wn * 32 + ni * 16 + l15;
                int e0 = kk + (laneHi ? 8 : 0);
                bfr[ni].q[0] = *(const uint4*)(&Bs[buf][col * 64 + e0]);
                bfr[ni].q[1] = *(const uint4*)(&Bs[buf][col * 64 + e0 + 16]);
            }
            #pragma unroll
            for (int mi = 0; mi < 2; mi++)
                #pragma unroll
                for (int ni = 0; ni < 2; ni++)
                    acc[mi][ni] = __builtin_amdgcn_wmma_f32_16x16x32_bf16(
                        false, af[mi].v, false, bfr[ni].v, (short)0,
                        acc[mi][ni], false, false);
        }
        async_fence();
        __syncthreads();
    }

    // ---- epilogue: bias + store. C layout: vgpr r, lane<16 -> M=r, N=lane;
    //      lane>=16 -> M=r+8, N=lane-16.
    #pragma unroll
    for (int ni = 0; ni < 2; ni++) {
        int gcol = bn0 + wn * 32 + ni * 16 + l15;
        if (MODE == 0) {
            int sel = gcol / 768, jj = gcol % 768;
            int h = jj / 64, dcol = jj % 64;
            const float* bias = (sel == 0) ? bias0 : (sel == 1) ? bias1 : bias2;
            uint16_t* dstB    = (sel == 0) ? Qb : (sel == 1) ? Kb : Vb;
            float bv = bias[jj];
            #pragma unroll
            for (int mi = 0; mi < 2; mi++)
                #pragma unroll
                for (int r = 0; r < 8; r++) {
                    int gm = bm0 + wm * 32 + mi * 16 + r + laneHi * 8;
                    if (gm < Mrows) {
                        int bb = gm / 577, t = gm % 577;
                        dstB[(((size_t)bb * H_ + h) * N_ + t) * D_ + dcol] =
                            f2bf(acc[mi][ni][r] + bv);
                    }
                }
        } else {
            float bv = bias0[gcol];
            #pragma unroll
            for (int mi = 0; mi < 2; mi++)
                #pragma unroll
                for (int r = 0; r < 8; r++) {
                    int gm = bm0 + wm * 32 + mi * 16 + r + laneHi * 8;
                    if (gm < Mrows)
                        Out[(size_t)gm * 768 + gcol] = acc[mi][ni][r] + bv;
                }
        }
    }
}

// ---------------------------------------------------------------------------
// RoPE: rows t>=1, pair (2p,2p+1):
//   out[2p+j] = pe[t-1,p,j,0]*x[2p] + pe[t-1,p,j,1]*x[2p+1]
// ---------------------------------------------------------------------------
__global__ void rope_kernel(uint16_t* __restrict__ Qb, uint16_t* __restrict__ Kb,
                            const float* __restrict__ pe) {
    int i = blockIdx.x * blockDim.x + threadIdx.x;
    const int total = B_ * H_ * (N_ - 1) * (D_ / 2);
    if (i >= total) return;
    int p  = i & 31;
    int t1 = (i >> 5) % (N_ - 1);
    int bh = i / ((N_ - 1) * (D_ / 2));
    size_t base = ((size_t)bh * N_ + (t1 + 1)) * D_ + 2 * p;
    const float* pp = pe + ((size_t)t1 * (D_ / 2) + p) * 4;
    float a00 = pp[0], a01 = pp[1], a10 = pp[2], a11 = pp[3];
    float q0 = bf2f(Qb[base]), q1 = bf2f(Qb[base + 1]);
    Qb[base]     = f2bf(a00 * q0 + a01 * q1);
    Qb[base + 1] = f2bf(a10 * q0 + a11 * q1);
    float k0 = bf2f(Kb[base]), k1 = bf2f(Kb[base + 1]);
    Kb[base]     = f2bf(a00 * k0 + a01 * k1);
    Kb[base + 1] = f2bf(a10 * k0 + a11 * k1);
}

// ---------------------------------------------------------------------------
// Flash attention: block = 8 waves on one (b,h); each wave owns 16 query
// rows; key/value streamed through LDS in 32-row blocks; online softmax.
// Writes O directly in [b, n, 768] bf16 layout for the output projection.
// ---------------------------------------------------------------------------
__global__ __launch_bounds__(256) void attn_kernel(
    const uint16_t* __restrict__ Qb, const uint16_t* __restrict__ Kb,
    const uint16_t* __restrict__ Vb, uint16_t* __restrict__ Ob) {
    __shared__ __align__(16) uint16_t Kt[32 * 64];    // K block, row-major
    __shared__ __align__(16) uint16_t VT[64 * 32];    // V block, transposed
    __shared__ __align__(16) uint16_t Ps[8][16 * 32]; // per-wave P scratch
    const int tid    = threadIdx.x;
    const int lane   = tid & 31;
    const int w      = tid >> 5;
    const int laneHi = lane >> 4, l15 = lane & 15;
    const int bh = blockIdx.x;
    const int q0 = blockIdx.y * 128 + w * 16;
    const int b = bh / H_, h = bh % H_;
    const uint16_t* Qp = Qb + (size_t)bh * N_ * D_;
    const uint16_t* Kp = Kb + (size_t)bh * N_ * D_;
    const uint16_t* Vp = Vb + (size_t)bh * N_ * D_;

    // Preload Q A-fragments for K-chunks {0..31},{32..63} of d=64.
    Frag qf[2];
    {
        int row = q0 + l15; if (row > N_ - 1) row = N_ - 1;
        #pragma unroll
        for (int kc = 0; kc < 2; kc++) {
            int e0 = kc * 32 + (laneHi ? 8 : 0);
            qf[kc].q[0] = *(const uint4*)(Qp + (size_t)row * D_ + e0);
            qf[kc].q[1] = *(const uint4*)(Qp + (size_t)row * D_ + e0 + 16);
        }
    }

    float m[8], l[8];
    #pragma unroll
    for (int r = 0; r < 8; r++) { m[r] = -3e38f; l[r] = 0.f; }
    v8f dacc[4] = {};
    const float scale = 0.03608439182435161f;   // 1/sqrt(768)
    const int nkb = (N_ + 31) / 32;             // 19 key blocks

    for (int kb = 0; kb < nkb; kb++) {
        int kr0 = kb * 32;
        {   // cooperative stage: K via async-to-LDS, V transposed by hand
            int row = tid >> 3, seg = tid & 7;
            int gr = kr0 + row; if (gr > N_ - 1) gr = N_ - 1;
            copy16(Kp + (size_t)gr * D_ + seg * 8, Kt + row * 64 + seg * 8);
            uint4 vv = *(const uint4*)(Vp + (size_t)gr * D_ + seg * 8);
            const uint16_t* ve = (const uint16_t*)&vv;
            #pragma unroll
            for (int e = 0; e < 8; e++) VT[(seg * 8 + e) * 32 + row] = ve[e];
        }
        if (kb + 1 < nkb) {
            __builtin_prefetch(Kp + (size_t)(kr0 + 32) * D_, 0, 1);
            __builtin_prefetch(Vp + (size_t)(kr0 + 32) * D_, 0, 1);
        }
        async_fence();
        __syncthreads();

        // S = Q K^T for two 16-col tiles (B kdim = d).
        v8f s[2] = {};
        #pragma unroll
        for (int kc = 0; kc < 2; kc++)
            #pragma unroll
            for (int j = 0; j < 2; j++) {
                Frag bfr;
                int off = (j * 16 + l15) * 64 + kc * 32 + (laneHi ? 8 : 0);
                bfr.q[0] = *(const uint4*)(Kt + off);
                bfr.q[1] = *(const uint4*)(Kt + off + 16);
                s[j] = __builtin_amdgcn_wmma_f32_16x16x32_bf16(
                    false, qf[kc].v, false, bfr.v, (short)0, s[j], false, false);
            }

        // scale + key-column mask
        #pragma unroll
        for (int j = 0; j < 2; j++) {
            bool valid = (kr0 + j * 16 + l15) < N_;
            #pragma unroll
            for (int r = 0; r < 8; r++)
                s[j][r] = valid ? s[j][r] * scale : -3e38f;
        }

        // online softmax: rows live across 16-lane halves of the C layout
        float corr[8];
        #pragma unroll
        for (int r = 0; r < 8; r++) {
            float mr = fmaxf(s[0][r], s[1][r]);
            mr = fmaxf(mr, __shfl_xor(mr, 1, 32));
            mr = fmaxf(mr, __shfl_xor(mr, 2, 32));
            mr = fmaxf(mr, __shfl_xor(mr, 4, 32));
            mr = fmaxf(mr, __shfl_xor(mr, 8, 32));
            float mn = fmaxf(m[r], mr);
            corr[r] = __expf(m[r] - mn);
            m[r] = mn;
            float p0 = __expf(s[0][r] - mn);
            float p1 = __expf(s[1][r] - mn);
            s[0][r] = p0; s[1][r] = p1;
            float rs = p0 + p1;
            rs += __shfl_xor(rs, 1, 32);
            rs += __shfl_xor(rs, 2, 32);
            rs += __shfl_xor(rs, 4, 32);
            rs += __shfl_xor(rs, 8, 32);
            l[r] = l[r] * corr[r] + rs;
        }

        // P (C layout) -> per-wave LDS scratch -> A fragment; rescale O acc
        #pragma unroll
        for (int r = 0; r < 8; r++) {
            int prow = r + laneHi * 8;
            Ps[w][prow * 32 + l15]      = f2bf(s[0][r]);
            Ps[w][prow * 32 + 16 + l15] = f2bf(s[1][r]);
            #pragma unroll
            for (int t = 0; t < 4; t++) dacc[t][r] *= corr[r];
        }
        asm volatile("s_wait_dscnt 0" ::: "memory");
        Frag pa;
        {
            int e0 = laneHi ? 8 : 0;
            pa.q[0] = *(const uint4*)(&Ps[w][l15 * 32 + e0]);
            pa.q[1] = *(const uint4*)(&Ps[w][l15 * 32 + 16 + e0]);
        }
        #pragma unroll
        for (int nt = 0; nt < 4; nt++) {
            Frag vbr;
            int off = (nt * 16 + l15) * 32 + (laneHi ? 8 : 0);
            vbr.q[0] = *(const uint4*)(VT + off);
            vbr.q[1] = *(const uint4*)(VT + off + 16);
            dacc[nt] = __builtin_amdgcn_wmma_f32_16x16x32_bf16(
                false, pa.v, false, vbr.v, (short)0, dacc[nt], false, false);
        }
        __syncthreads();
    }

    // epilogue: normalize by row sums, store bf16 into [b, n, 768]
    #pragma unroll
    for (int r = 0; r < 8; r++) {
        int t = q0 + r + laneHi * 8;
        if (t < N_) {
            float inv = 1.0f / l[r];
            size_t rowbase = ((size_t)b * N_ + t) * E_ + h * D_;
            #pragma unroll
            for (int nt = 0; nt < 4; nt++)
                Ob[rowbase + nt * 16 + l15] = f2bf(dacc[nt][r] * inv);
        }
    }
}

// ---------------------------------------------------------------------------
extern "C" void kernel_launch(void* const* d_in, const int* in_sizes, int n_in,
                              void* d_out, int out_size, void* d_ws,
                              size_t ws_size, hipStream_t stream) {
    const float* x  = (const float*)d_in[0];
    const float* pe = (const float*)d_in[1];
    const float* Wq = (const float*)d_in[2];
    const float* bq = (const float*)d_in[3];
    const float* Wk = (const float*)d_in[4];
    const float* bk = (const float*)d_in[5];
    const float* Wv = (const float*)d_in[6];
    const float* bv = (const float*)d_in[7];
    const float* Wp = (const float*)d_in[8];
    const float* bp = (const float*)d_in[9];
    float* out = (float*)d_out;

    char* ws = (char*)d_ws;
    size_t off = 0;
    auto alloc = [&](size_t bytes) {
        size_t o = off;
        off += (bytes + 255) & ~(size_t)255;
        return o;
    };
    const size_t tensorBytes = (size_t)M_ * E_ * 2;        // 28,360,704 B
    uint16_t* xb   = (uint16_t*)(ws + alloc(tensorBytes));
    uint16_t* wqkv = (uint16_t*)(ws + alloc((size_t)3 * E_ * E_ * 2));
    uint16_t* wpb  = (uint16_t*)(ws + alloc((size_t)E_ * E_ * 2));
    uint16_t* QB   = (uint16_t*)(ws + alloc(tensorBytes));
    uint16_t* KB   = (uint16_t*)(ws + alloc(tensorBytes));
    uint16_t* VB   = (uint16_t*)(ws + alloc(tensorBytes));
    uint16_t* OB   = xb;   // x-bf16 dead after QKV GEMM; reuse for O

    const int nX = M_ * E_;           // 14,180,352
    const int nW = E_ * E_;           // 589,824
    f32_to_bf16_kernel<<<(nX + 255) / 256, 256, 0, stream>>>(x, xb, nX);
    f32_to_bf16_kernel<<<(nW + 255) / 256, 256, 0, stream>>>(Wq, wqkv, nW);
    f32_to_bf16_kernel<<<(nW + 255) / 256, 256, 0, stream>>>(Wk, wqkv + nW, nW);
    f32_to_bf16_kernel<<<(nW + 255) / 256, 256, 0, stream>>>(Wv, wqkv + 2 * nW, nW);
    f32_to_bf16_kernel<<<(nW + 255) / 256, 256, 0, stream>>>(Wp, wpb, nW);

    // Fused QKV projection: M=18464, N=2304, K=768
    gemm_bf16_kernel<0><<<dim3((M_ + 127) / 128, 2304 / 64), 256, 0, stream>>>(
        xb, wqkv, bq, bk, bv, QB, KB, VB, nullptr, M_);

    // RoPE on Q,K
    const int nR = B_ * H_ * (N_ - 1) * (D_ / 2);   // 7,077,888
    rope_kernel<<<(nR + 255) / 256, 256, 0, stream>>>(QB, KB, pe);

    // Flash attention: (b*h) x query blocks of 128
    attn_kernel<<<dim3(B_ * H_, (N_ + 127) / 128), 256, 0, stream>>>(QB, KB, VB, OB);

    // Output projection: M=18464, N=768, K=768 -> fp32 d_out
    gemm_bf16_kernel<1><<<dim3((M_ + 127) / 128, 768 / 64), 256, 0, stream>>>(
        OB, wpb, bp, nullptr, nullptr, nullptr, nullptr, nullptr, out, M_);
}
